// GTCN2_14491219657208
// MI455X (gfx1250) — matrix operand order repaired and checked by reading, not compiled, not run on hardware
//
#include <hip/hip_runtime.h>
#include <hip/hip_bf16.h>

typedef __attribute__((ext_vector_type(16))) __bf16 v16bf;
typedef __attribute__((ext_vector_type(8)))  float  v8f;

// ---------------------------------------------------------------------------
// Pack an f32 weight matrix W[K][Nw] (row-major) into bf16, fragment-ordered:
// Wpk[((ntile*KSTEPS + ks)*32 + lane)*16 + j] = W[ks*32 + (lane>>4)*16 + j][ntile*16 + (lane&15)]
// so that each lane's WMMA B-fragment is one contiguous 32-byte load.
// ---------------------------------------------------------------------------
__global__ void pack_weight(const float* __restrict__ W, __bf16* __restrict__ Wpk,
                            int K, int Nw) {
    int idx = blockIdx.x * blockDim.x + threadIdx.x;
    int total = K * Nw;
    if (idx >= total) return;
    int ksteps = K >> 5;
    int j    = idx & 15;
    int lane = (idx >> 4) & 31;
    int t    = idx >> 9;               // = ntile*ksteps + ks
    int ks   = t % ksteps;
    int tile = t / ksteps;
    int n = tile * 16 + (lane & 15);
    int k = ks * 32 + (lane >> 4) * 16 + j;
    Wpk[idx] = (__bf16)W[(size_t)k * Nw + n];
}

// ---------------------------------------------------------------------------
// t[i] = A2[row] * x0[i]   (row = i / 256)
// ---------------------------------------------------------------------------
__global__ void init_t(const float* __restrict__ A2, const float* __restrict__ x0,
                       float* __restrict__ t, int total) {
    int i = blockIdx.x * blockDim.x + threadIdx.x;
    if (i < total) t[i] = A2[i >> 8] * x0[i];
}

// ---------------------------------------------------------------------------
// SpMM: one wave per edge. t[row,:] += val * h[col,:] via f32 global atomics.
// 32 lanes x 8 rounds cover F=256; every round is a coalesced 128B burst.
// ---------------------------------------------------------------------------
__global__ void spmm_edges(const int* __restrict__ rows, const int* __restrict__ cols,
                           const float* __restrict__ vals, const float* __restrict__ h,
                           float* __restrict__ t, int E) {
    int gw   = (blockIdx.x * blockDim.x + threadIdx.x) >> 5;
    int lane = threadIdx.x & 31;
    if (gw >= E) return;
    int r = rows[gw];
    int c = cols[gw];
    float v = vals[gw];
    const float* src = h + (size_t)c * 256;
    float*       dst = t + (size_t)r * 256;
#pragma unroll
    for (int i = 0; i < 8; ++i) {
        int f = lane + 32 * i;
        atomicAdd(&dst[f], v * src[f]);
    }
}

// ---------------------------------------------------------------------------
// WMMA GEMM: out[N, NW] = act( A[N, K] @ Wpk + bias (+ addsrc) )
//  - A is f32 in global; one 16-row M-tile per block staged to LDS as bf16.
//  - each wave computes one 16x16 output tile, K in steps of 32 via
//    v_wmma_f32_16x16x32_bf16 (f32 accumulate).
//  - blockDim.x = 32 * (NW / 16 / gridDim.x) waves.
// ---------------------------------------------------------------------------
template <int K, int NW, int RELU, int HASADD>
__global__ void gemm_wmma(const float* __restrict__ A, const __bf16* __restrict__ Wpk,
                          const float* __restrict__ bias, const float* __restrict__ addsrc,
                          float* __restrict__ out) {
    constexpr int KSTEPS = K / 32;
    __shared__ __bf16 lds[16 * K];

    const int m0     = blockIdx.y * 16;
    const int waves  = blockDim.x >> 5;
    const int waveId = threadIdx.x >> 5;
    const int lane   = threadIdx.x & 31;
    const int ntile  = blockIdx.x * waves + waveId;
    const int n0     = ntile * 16;

    // Stage the 16 x K f32 activation tile into LDS as bf16 (shared by all waves).
    for (int i = threadIdx.x; i < 16 * K; i += blockDim.x) {
        int r = i / K;
        int c = i % K;
        lds[i] = (__bf16)A[(size_t)(m0 + r) * K + c];
    }
    __syncthreads();

    const int half = lane >> 4;      // 0: lanes 0-15, 1: lanes 16-31
    const int mloc = lane & 15;      // A-fragment row for this lane

    v8f acc = {};
#pragma unroll
    for (int ks = 0; ks < KSTEPS; ++ks) {
        // A fragment: 16x32 bf16, lane holds row mloc; VGPR v holds
        // k = ks*32 + half*16 + v*2 (+1), matching packed B.
        v16bf afrag;
#pragma unroll
        for (int v = 0; v < 8; ++v) {
            int kk = ks * 32 + (half << 4) + (v << 1);
            afrag[2 * v]     = lds[mloc * K + kk];
            afrag[2 * v + 1] = lds[mloc * K + kk + 1];
        }
        // B fragment: single contiguous 32-byte load from packed weights.
        v16bf bfrag = *(const v16bf*)(Wpk + ((size_t)(ntile * KSTEPS + ks) * 32 + lane) * 16);

        acc = __builtin_amdgcn_wmma_f32_16x16x32_bf16(
            /*neg_a=*/false, afrag, /*neg_b=*/false, bfrag,
            /*c_mod=*/(short)0, acc, /*reuse_a=*/false, /*reuse_b=*/false);
    }

    // C/D layout: VGPR v -> M = v + 8*half ; lane -> N = lane & 15.
    const int col = n0 + (lane & 15);
    const float b = bias[col];
#pragma unroll
    for (int v = 0; v < 8; ++v) {
        int row = m0 + v + 8 * half;
        float val = acc[v] + b;
        if (HASADD) val += addsrc[(size_t)row * NW + col];
        if (RELU)   val = fmaxf(val, 0.0f);
        out[(size_t)row * NW + col] = val;
    }
}

// ---------------------------------------------------------------------------
// Host-side orchestration (graph-capture safe: launches only).
// ---------------------------------------------------------------------------
extern "C" void kernel_launch(void* const* d_in, const int* in_sizes, int n_in,
                              void* d_out, int out_size, void* d_ws, size_t ws_size,
                              hipStream_t stream) {
    const float* x     = (const float*)d_in[0];
    const int*   erow  = (const int*)d_in[1];
    const int*   ecol  = (const int*)d_in[2];
    const float* evals = (const float*)d_in[3];
    const float* A2    = (const float*)d_in[4];
    const float* W1    = (const float*)d_in[5];
    const float* b1    = (const float*)d_in[6];
    const float* W3    = (const float*)d_in[7];
    const float* b3    = (const float*)d_in[8];
    const float* W2    = (const float*)d_in[9];
    const float* b2    = (const float*)d_in[10];
    float* out = (float*)d_out;

    const int N = 100000;
    const int E = 3200000;
    const size_t feat = (size_t)N * 256;

    float* x0 = (float*)d_ws;
    float* t  = x0 + feat;
    float* h  = t + feat;
    __bf16* W1p = (__bf16*)(h + feat);
    __bf16* W3p = (__bf16*)((char*)W1p + (size_t)512 * 256 * 2);
    __bf16* W2p = (__bf16*)((char*)W3p + (size_t)256 * 256 * 2);

    // One-time weight conversion to fragment-ordered bf16.
    pack_weight<<<(512 * 256 + 255) / 256, 256, 0, stream>>>(W1, W1p, 512, 256);
    pack_weight<<<(256 * 256 + 255) / 256, 256, 0, stream>>>(W3, W3p, 256, 256);
    pack_weight<<<(256 * 64 + 255) / 256, 256, 0, stream>>>(W2, W2p, 256, 64);

    // x0 = relu(x @ W1 + b1): K=512, NW=256, 8 waves/block, 2 N-blocks.
    gemm_wmma<512, 256, 1, 0><<<dim3(2, N / 16), 256, 0, stream>>>(x, W1p, b1, nullptr, x0);

    const float* hcur = x0;
    for (int hop = 0; hop < 10; ++hop) {
        // t = A2 * x0
        init_t<<<(int)(feat / 256), 256, 0, stream>>>(A2, x0, t, (int)feat);
        // t += spmm(A, hcur)
        spmm_edges<<<(E * 32) / 256, 256, 0, stream>>>(erow, ecol, evals, hcur, t, E);
        // h = relu(t + t @ W3 + b3)
        gemm_wmma<256, 256, 1, 1><<<dim3(2, N / 16), 256, 0, stream>>>(t, W3p, b3, t, h);
        hcur = h;
    }

    // out = h @ W2 + b2: NW=64, 4 waves/block.
    gemm_wmma<256, 64, 0, 0><<<dim3(1, N / 16), 128, 0, stream>>>(h, W2p, b2, nullptr, out);
}
